// VisionMambaBlock_1657857376473
// MI455X (gfx1250) — matrix operand order repaired and checked by reading, not compiled, not run on hardware
//
#include <hip/hip_runtime.h>

// ---- problem constants ----
#define DIM_C    64
#define DINNER   128
#define SEQ_L    110592            // 48*48*48
#define NBATCH   2
#define TOT_BL   (NBATCH * SEQ_L)  // 221184 rows
#define CHUNK    512
#define NCHUNK   (SEQ_L / CHUNK)   // 216

typedef __attribute__((ext_vector_type(16))) _Float16 v16h;
typedef __attribute__((ext_vector_type(8)))  _Float16 v8h;
typedef __attribute__((ext_vector_type(8)))  float    v8f;
typedef __attribute__((ext_vector_type(4)))  float    v4f;
typedef __attribute__((ext_vector_type(4)))  unsigned int u32x4;
typedef __attribute__((ext_vector_type(8)))  unsigned int u32x8;

static __device__ __forceinline__ float siluf(float v) {
  return v / (1.0f + __expf(-v));
}
static __device__ __forceinline__ float softplusf(float v) {
  return fmaxf(v, 0.0f) + __logf(1.0f + __expf(-fabsf(v)));
}

// ---- Tensor Data Mover: 2D f16 tile, global -> LDS (ISA 08_async_tensor §8) ----
// Issued once per wave (EXEC ignored); caller gates to one wave and waits TENSORcnt.
static __device__ __forceinline__ void tdm_load_2d_f16(
    unsigned lds_off, const void* gptr,
    unsigned dim0, unsigned dim1, unsigned tile0, unsigned tile1, unsigned stride0)
{
  unsigned long long ga = (unsigned long long)gptr;
  u32x4 g0;
  g0[0] = 1u;                                            // count=1, is_restore=0
  g0[1] = lds_off;                                       // lds_addr (bytes)
  g0[2] = (unsigned)(ga & 0xffffffffu);                  // global_addr[31:0]
  g0[3] = (unsigned)((ga >> 32) & 0x01ffffffu) | 0x80000000u; // addr[56:32] | type=2
  u32x8 g1;
  g1[0] = 1u << 16;                                      // data_size=1 (2B), wg_mask=0
  g1[1] = (dim0 & 0xffffu) << 16;                        // tensor_dim0[15:0] @ bits63:48
  g1[2] = ((dim0 >> 16) & 0xffffu) | ((dim1 & 0xffffu) << 16); // dim0 hi | dim1 lo
  g1[3] = ((dim1 >> 16) & 0xffffu) | (tile0 << 16);      // dim1 hi | tile_dim0
  g1[4] = (tile1 & 0xffffu);                             // tile_dim1 (tile_dim2=0)
  g1[5] = stride0;                                       // tensor_dim0_stride[31:0]
  g1[6] = 0u;                                            // stride0 hi | stride1 lo
  g1[7] = 0u;                                            // stride1 hi
  asm volatile("tensor_load_to_lds %0, %1" :: "s"(g0), "s"(g1) : "memory");
}

// ---- async global -> LDS copy (16B per lane), tracked by ASYNCcnt ----
static __device__ __forceinline__ void async_copy_b128(unsigned lds_off, const void* g) {
  unsigned long long ga = (unsigned long long)g;
  asm volatile("global_load_async_to_lds_b128 %0, %1, off" :: "v"(lds_off), "v"(ga) : "memory");
}
static __device__ __forceinline__ void wait_async0() {
  asm volatile("s_wait_asynccnt 0x0" ::: "memory");
}

// ---- WMMA fragment gathers (CDNA5 16-bit layouts, wave32) ----
static __device__ __forceinline__ v16h load_a_frag(const _Float16* rowbase, int k0, int lane) {
  const int kb = (lane >= 16) ? 8 : 0;
  v8h lo = *reinterpret_cast<const v8h*>(rowbase + k0 + kb);
  v8h hi = *reinterpret_cast<const v8h*>(rowbase + k0 + 16 + kb);
  v16h a;
#pragma unroll
  for (int i = 0; i < 8; ++i) { a[i] = lo[i]; a[i + 8] = hi[i]; }
  return a;
}
// A-frag from an f32 LDS row, converting to f16 at gather time
static __device__ __forceinline__ v16h load_a_frag_f32(const float* rowbase, int k0, int lane) {
  const int kb = (lane >= 16) ? 8 : 0;
  v4f l0 = *reinterpret_cast<const v4f*>(rowbase + k0 + kb);
  v4f l1 = *reinterpret_cast<const v4f*>(rowbase + k0 + kb + 4);
  v4f h0 = *reinterpret_cast<const v4f*>(rowbase + k0 + 16 + kb);
  v4f h1 = *reinterpret_cast<const v4f*>(rowbase + k0 + 16 + kb + 4);
  v16h a;
#pragma unroll
  for (int i = 0; i < 4; ++i) {
    a[i]      = (_Float16)l0[i];
    a[i + 4]  = (_Float16)l1[i];
    a[i + 8]  = (_Float16)h0[i];
    a[i + 12] = (_Float16)h1[i];
  }
  return a;
}
static __device__ __forceinline__ v16h load_b_frag(const _Float16* colbase, int k0, int lane) {
  const int kb = (lane >= 16) ? 8 : 0;
  v8h lo = *reinterpret_cast<const v8h*>(colbase + k0 + kb);
  v8h hi = *reinterpret_cast<const v8h*>(colbase + k0 + 16 + kb);
  v16h b;
#pragma unroll
  for (int i = 0; i < 8; ++i) { b[i] = lo[i]; b[i + 8] = hi[i]; }
  return b;
}

// ============ kernel 0: weight conversion to f16, column-major [n][k] ============
// WcatH column-tiles are interleaved: tile 2j = W_x[:, 16j..16j+15], tile 2j+1 =
// W_dt[:, 16j..16j+15] so one wave owning 8 consecutive tiles holds 4 (xp,dt) pairs.
__global__ void cvt_weights(const float* __restrict__ W_in,
                            const float* __restrict__ W_x,
                            const float* __restrict__ W_dt,
                            const float* __restrict__ W_out,
                            _Float16* __restrict__ WinH,      // [256][64]
                            _Float16* __restrict__ WcatH,     // [256][128] interleaved
                            _Float16* __restrict__ WoutH)     // [64][128]
{
  const int stride = gridDim.x * blockDim.x;
  const int tid = blockIdx.x * blockDim.x + threadIdx.x;
  for (int i = tid; i < 256 * 64; i += stride) {
    int n = i >> 6, k = i & 63;
    WinH[i] = (_Float16)W_in[k * 256 + n];
  }
  for (int i = tid; i < 256 * 128; i += stride) {
    int n = i >> 7, k = i & 127;
    int g = n >> 4, cl = n & 15;
    int srcc = (g >> 1) * 16 + cl;
    float v = (g & 1) ? W_dt[k * 128 + srcc] : W_x[k * 144 + srcc];
    WcatH[i] = (_Float16)v;
  }
  for (int i = tid; i < 64 * 128; i += stride) {
    int n = i >> 7, k = i & 127;
    WoutH[i] = (_Float16)W_out[k * 64 + n];
  }
}

// ============ kernel 1: x[BL,64] @ W_in[64,256] -> xssm|res (f16) ============
// 8 waves; block = 64 rows x 256 cols. Wave = (row-strip = w>>1, col-half = w&1);
// each wave: 16 rows x 8 n-tiles -> 64 accumulator VGPRs (spill-free).
__global__ void gemm_in(const float* __restrict__ X,
                        const _Float16* __restrict__ Bh,      // [256][64]
                        _Float16* __restrict__ xssm,
                        _Float16* __restrict__ res)
{
  extern __shared__ char smem[];
  _Float16* ldsB = (_Float16*)smem;           // 256*64 h = 32 KB @ LDS 0
  float*    ldsA = (float*)(ldsB + 256 * 64); // 64*64 f32 = 16 KB @ LDS 32768
  const int tid = threadIdx.x;
  const int lane = tid & 31;
  const int wave = tid >> 5;
  const size_t row0 = (size_t)blockIdx.x * 64;

  if (wave == 0) {
    tdm_load_2d_f16(0u, Bh, 64u, 256u, 64u, 256u, 64u);
  }
  const char* asrc = (const char*)(X + row0 * 64);   // 16 KB contiguous
#pragma unroll
  for (int it = 0; it < 4; ++it) {
    unsigned off = (unsigned)(tid + it * 256) * 16u;
    async_copy_b128(32768u + off, asrc + off);
  }
  wait_async0();
  if (wave == 0) __builtin_amdgcn_s_wait_tensorcnt(0);
  __syncthreads();

  const int rs = wave >> 1;                 // row strip 0..3
  const int h  = wave & 1;                  // column half
  v8f acc[8];
#pragma unroll
  for (int t = 0; t < 8; ++t) acc[t] = v8f{};
  const float* arow = ldsA + (rs * 16 + (lane & 15)) * 64;
#pragma unroll
  for (int k0 = 0; k0 < 64; k0 += 32) {
    v16h a = load_a_frag_f32(arow, k0, lane);
#pragma unroll
    for (int t = 0; t < 8; ++t) {
      v16h b = load_b_frag(ldsB + ((h * 8 + t) * 16 + (lane & 15)) * 64, k0, lane);
      acc[t] = __builtin_amdgcn_wmma_f32_16x16x32_f16(false, a, false, b, (short)0, acc[t], false, false);
    }
  }
  // tiles 0..7 (h=0) -> xssm cols; tiles 8..15 (h=1) -> res cols
  _Float16* dst = h ? res : xssm;
  const int colb = lane & 15;
  const int mhi = (lane >= 16) ? 8 : 0;
#pragma unroll
  for (int t = 0; t < 8; ++t) {
    const int col = t * 16 + colb;
#pragma unroll
    for (int r = 0; r < 8; ++r) {
      size_t row = row0 + rs * 16 + r + mhi;
      dst[row * DINNER + col] = (_Float16)acc[t][r];
    }
  }
}

// ============ kernel 2: depthwise conv1d(k=4, pad 1/2) + bias + SiLU ============
__global__ void dwconv_silu(const _Float16* __restrict__ xssm,
                            const float* __restrict__ cw,     // [4,1,128]
                            const float* __restrict__ cb,     // [128]
                            _Float16* __restrict__ U)
{
  size_t idx = (size_t)blockIdx.x * 256 + threadIdx.x;  // over BL*16 groups
  const int c0 = (int)(idx & 15) * 8;
  size_t pos = idx >> 4;
  int b = (pos >= SEQ_L) ? 1 : 0;
  int l = (int)(pos - (size_t)b * SEQ_L);
  float acc[8];
#pragma unroll
  for (int u = 0; u < 8; ++u) acc[u] = cb[c0 + u];
#pragma unroll
  for (int j = 0; j < 4; ++j) {
    int li = l - 1 + j;
    if (li >= 0 && li < SEQ_L) {
      v8h v = *reinterpret_cast<const v8h*>(xssm + ((size_t)b * SEQ_L + li) * DINNER + c0);
#pragma unroll
      for (int u = 0; u < 8; ++u) acc[u] += cw[j * 128 + c0 + u] * (float)v[u];
    }
  }
  v8h o;
#pragma unroll
  for (int u = 0; u < 8; ++u) o[u] = (_Float16)siluf(acc[u]);
  *reinterpret_cast<v8h*>(U + pos * DINNER + c0) = o;
}

// ============ kernel 3: u[BL,128] @ Wcat (interleaved); p = xp*softplus(dt+b) ============
// block = 64 rows x 256 cols; wave = (row-strip, col-half); 8 tiles = 4 (xp,dt) pairs.
__global__ void gemm_ssm(const _Float16* __restrict__ U,
                         const _Float16* __restrict__ Bh,     // [256][128] interleaved
                         const float* __restrict__ b_dt,
                         float* __restrict__ P)               // [BL,128] f32
{
  extern __shared__ char smem[];
  _Float16* ldsB = (_Float16*)smem;            // 256*128 h = 64 KB @ LDS 0
  _Float16* ldsA = ldsB + 256 * 128;           // 64*128 h  = 16 KB @ LDS 65536
  const int tid = threadIdx.x;
  const int lane = tid & 31;
  const int wave = tid >> 5;
  const size_t row0 = (size_t)blockIdx.x * 64;

  if (wave == 0) {
    tdm_load_2d_f16(0u, Bh, 128u, 256u, 128u, 256u, 128u);
  }
  const char* asrc = (const char*)(U + row0 * DINNER);   // 16 KB contiguous
#pragma unroll
  for (int it = 0; it < 4; ++it) {
    unsigned off = (unsigned)(tid + it * 256) * 16u;
    async_copy_b128(65536u + off, asrc + off);
  }
  wait_async0();
  if (wave == 0) __builtin_amdgcn_s_wait_tensorcnt(0);
  __syncthreads();

  const int rs = wave >> 1;
  const int h  = wave & 1;
  const int gbase = h * 8;
  v8f acc[8];
#pragma unroll
  for (int t = 0; t < 8; ++t) acc[t] = v8f{};
  const _Float16* arow = ldsA + (rs * 16 + (lane & 15)) * 128;
#pragma unroll
  for (int k0 = 0; k0 < 128; k0 += 32) {
    v16h a = load_a_frag(arow, k0, lane);
#pragma unroll
    for (int t = 0; t < 8; ++t) {
      v16h b = load_b_frag(ldsB + ((gbase + t) * 16 + (lane & 15)) * 128, k0, lane);
      acc[t] = __builtin_amdgcn_wmma_f32_16x16x32_f16(false, a, false, b, (short)0, acc[t], false, false);
    }
  }
  // pairs: acc[2i] = x_proj cols, acc[2i+1] = dt cols; both map to col (h*4+i)*16+colb
  const int colb = lane & 15;
  const int mhi = (lane >= 16) ? 8 : 0;
#pragma unroll
  for (int i = 0; i < 4; ++i) {
    const int col = (h * 4 + i) * 16 + colb;
    const float bd = b_dt[col];
#pragma unroll
    for (int r = 0; r < 8; ++r) {
      float dt = softplusf(acc[2 * i + 1][r] + bd);
      P[(row0 + rs * 16 + r + mhi) * DINNER + col] = acc[2 * i][r] * dt;
    }
  }
}

// ============ kernel 4: per-chunk inclusive cumsum in place + chunk totals ============
__global__ void scan_chunks(float* __restrict__ P, float* __restrict__ csum)
{
  const int b  = blockIdx.x / NCHUNK;
  const int ch = blockIdx.x % NCHUNK;
  const int c  = threadIdx.x;                // 128 threads = channels
  size_t base = ((size_t)b * SEQ_L + (size_t)ch * CHUNK) * DINNER + c;
  float run = 0.0f;
  for (int i = 0; i < CHUNK; i += 4) {       // batched loads, short add chain
    float p0 = P[base + (size_t)(i + 0) * DINNER];
    float p1 = P[base + (size_t)(i + 1) * DINNER];
    float p2 = P[base + (size_t)(i + 2) * DINNER];
    float p3 = P[base + (size_t)(i + 3) * DINNER];
    float s0 = run + p0, s1 = s0 + p1, s2 = s1 + p2, s3 = s2 + p3;
    P[base + (size_t)(i + 0) * DINNER] = s0;
    P[base + (size_t)(i + 1) * DINNER] = s1;
    P[base + (size_t)(i + 2) * DINNER] = s2;
    P[base + (size_t)(i + 3) * DINNER] = s3;
    run = s3;
  }
  csum[((size_t)b * NCHUNK + ch) * DINNER + c] = run;
}

// ============ kernel 5: exclusive scan of chunk totals ============
__global__ void scan_totals(const float* __restrict__ csum, float* __restrict__ coff)
{
  const int t = threadIdx.x;                 // 256 = NBATCH*128
  const int b = t >> 7, c = t & 127;
  float run = 0.0f;
  for (int ch = 0; ch < NCHUNK; ++ch) {
    size_t i = ((size_t)b * NCHUNK + ch) * DINNER + c;
    coff[i] = run;
    run += csum[i];
  }
}

// ============ kernel 6: y=(p+off)*silu(res); out = y@W_out + x; LayerNorm ============
__global__ void gemm_out_ln(const float* __restrict__ P,
                            const float* __restrict__ coff,
                            const _Float16* __restrict__ Res,
                            const _Float16* __restrict__ Bh,  // [64][128]
                            const float* __restrict__ Xin,    // [BL,64]
                            const float* __restrict__ gamma,
                            const float* __restrict__ beta,
                            float* __restrict__ Out)          // [BL,64]
{
  extern __shared__ char smem[];
  _Float16* ldsB  = (_Float16*)smem;             // 64*128 h  = 16 KB @ LDS 0
  _Float16* ldsA  = ldsB + 64 * 128;             // 128*128 h = 32 KB
  float*    lnbuf = (float*)(ldsA + 128 * 128);  // 128*64 f32 = 32 KB
  const int tid = threadIdx.x;
  const int lane = tid & 31;
  const int wave = tid >> 5;
  const size_t row0 = (size_t)blockIdx.x * 128;

  if (wave == 0) {
    tdm_load_2d_f16(0u, Bh, 128u, 64u, 128u, 64u, 128u);
  }
  // build A = (p + chunk_offset) * silu(res) on the fly, store f16 (computed, no DMA)
  for (int i = tid; i < 128 * 128; i += 256) {
    int r = i >> 7, c = i & 127;
    size_t grow = row0 + r;
    int b = (grow >= SEQ_L) ? 1 : 0;
    int l = (int)(grow - (size_t)b * SEQ_L);
    int chnk = l / CHUNK;
    float off = coff[((size_t)b * NCHUNK + chnk) * DINNER + c];
    float st  = P[grow * DINNER + c] + off;
    float rs  = (float)Res[grow * DINNER + c];
    ldsA[i] = (_Float16)(st * siluf(rs));
  }
  if (wave == 0) __builtin_amdgcn_s_wait_tensorcnt(0);
  __syncthreads();

  v8f acc[4] = {{}, {}, {}, {}};
  const _Float16* arow = ldsA + (wave * 16 + (lane & 15)) * 128;
#pragma unroll
  for (int k0 = 0; k0 < 128; k0 += 32) {
    v16h a = load_a_frag(arow, k0, lane);
#pragma unroll
    for (int t = 0; t < 4; ++t) {
      v16h bf = load_b_frag(ldsB + (t * 16 + (lane & 15)) * 128, k0, lane);
      acc[t] = __builtin_amdgcn_wmma_f32_16x16x32_f16(false, a, false, bf, (short)0, acc[t], false, false);
    }
  }
  const int mhi = (lane >= 16) ? 8 : 0;
#pragma unroll
  for (int t = 0; t < 4; ++t) {
    int col = t * 16 + (lane & 15);
#pragma unroll
    for (int r = 0; r < 8; ++r) {
      int lrow = wave * 16 + r + mhi;
      float v = acc[t][r] + Xin[(row0 + lrow) * DIM_C + col];
      lnbuf[lrow * DIM_C + col] = v;
    }
  }
  __syncthreads();
  if (tid < 128) {
    const float* z = lnbuf + tid * DIM_C;
    float mu = 0.0f;
#pragma unroll
    for (int c = 0; c < DIM_C; ++c) mu += z[c];
    mu *= (1.0f / DIM_C);
    float var = 0.0f;
#pragma unroll
    for (int c = 0; c < DIM_C; ++c) { float d = z[c] - mu; var += d * d; }
    var *= (1.0f / DIM_C);
    float inv = rsqrtf(var + 1e-3f);
    size_t orow = (row0 + tid) * DIM_C;
#pragma unroll
    for (int c = 0; c < DIM_C; ++c)
      Out[orow + c] = gamma[c] * (z[c] - mu) * inv + beta[c];
  }
}

// ============ host launcher ============
extern "C" void kernel_launch(void* const* d_in, const int* in_sizes, int n_in,
                              void* d_out, int out_size, void* d_ws, size_t ws_size,
                              hipStream_t stream) {
  const float* x      = (const float*)d_in[0];
  const float* W_in   = (const float*)d_in[1];
  const float* conv_w = (const float*)d_in[2];
  const float* conv_b = (const float*)d_in[3];
  const float* W_x    = (const float*)d_in[4];
  const float* W_dt   = (const float*)d_in[5];
  const float* b_dt   = (const float*)d_in[6];
  const float* W_out  = (const float*)d_in[7];
  const float* gamma  = (const float*)d_in[8];
  const float* beta   = (const float*)d_in[9];
  float* out = (float*)d_out;

  char* ws = (char*)d_ws;
  size_t off = 0;
  auto alloc = [&](size_t bytes) -> char* {
    char* p = ws + off;
    off += (bytes + 255) & ~(size_t)255;
    return p;
  };
  _Float16* xssm = (_Float16*)alloc((size_t)TOT_BL * DINNER * 2);
  _Float16* resh = (_Float16*)alloc((size_t)TOT_BL * DINNER * 2);
  _Float16* uh   = (_Float16*)alloc((size_t)TOT_BL * DINNER * 2);
  float*    P    = (float*)   alloc((size_t)TOT_BL * DINNER * 4);
  float*    csum = (float*)   alloc((size_t)NBATCH * NCHUNK * DINNER * 4);
  float*    coff = (float*)   alloc((size_t)NBATCH * NCHUNK * DINNER * 4);
  _Float16* WinH  = (_Float16*)alloc(256 * 64 * 2);
  _Float16* WcatH = (_Float16*)alloc(256 * 128 * 2);
  _Float16* WoutH = (_Float16*)alloc(64 * 128 * 2);

  cvt_weights<<<dim3(64), dim3(256), 0, stream>>>(W_in, W_x, W_dt, W_out, WinH, WcatH, WoutH);

  const size_t smem1 = 256 * 64 * 2 + 64 * 64 * 4;                  // 48 KB
  gemm_in<<<dim3(TOT_BL / 64), dim3(256), smem1, stream>>>(x, WinH, xssm, resh);

  dwconv_silu<<<dim3((TOT_BL * 16) / 256), dim3(256), 0, stream>>>(xssm, conv_w, conv_b, uh);

  const size_t smem2 = 256 * 128 * 2 + 64 * 128 * 2;                // 80 KB
  gemm_ssm<<<dim3(TOT_BL / 64), dim3(256), smem2, stream>>>(uh, WcatH, b_dt, P);

  scan_chunks<<<dim3(NBATCH * NCHUNK), dim3(128), 0, stream>>>(P, csum);
  scan_totals<<<dim3(1), dim3(256), 0, stream>>>(csum, coff);

  const size_t smem3 = 64 * 128 * 2 + 128 * 128 * 2 + 128 * 64 * 4; // 80 KB
  gemm_out_ln<<<dim3(TOT_BL / 128), dim3(256), smem3, stream>>>(P, coff, resh, WoutH, x, gamma, beta, out);
}
